// PointMLPBackbone_30511447671615
// MI455X (gfx1250) — matrix operand order repaired and checked by reading, not compiled, not run on hardware
//
#include <hip/hip_runtime.h>
#include <hip/hip_bf16.h>
#include <math.h>

typedef __attribute__((ext_vector_type(16))) _Float16 v16h;
typedef __attribute__((ext_vector_type(8)))  _Float16 v8h;
typedef __attribute__((ext_vector_type(4)))  _Float16 v4h;
typedef __attribute__((ext_vector_type(8)))  float    v8f;

#define KNN 24

// ---------------------------------------------------------------------------
// Batched-column GEMM:  Y[bb,o,j] = act( sum_i W[o,i] * X[bb,i,j] + b[o] (+R) )
// X: (NB, Cin, Lb) row-major, Y/R: (NB, Cout, Lb).
// REQUIRES: Cout % 64 == 0, Cin % 32 == 0 (true for all launches here).
// Block: 256 threads = 8 waves; tile 64(M) x 64(N); each wave owns one 16-row
// M sub-tile and two 16-col N sub-tiles -> shared A fragment, 2 WMMAs/K-step.
// LDS pitch = 40 halves (80B, 16B multiple) -> conflict-free ds_load_b128.
// ---------------------------------------------------------------------------
__global__ __launch_bounds__(256)
void gemm_wmma_kernel(const float* __restrict__ W, const float* __restrict__ bias,
                      const float* __restrict__ X, const float* __restrict__ R,
                      float* __restrict__ Y,
                      int Cout, int Cin, int Lb, int NB, int relu) {
    const int L  = NB * Lb;
    const int n0 = blockIdx.x * 64;
    const int m0 = blockIdx.y * 64;

    __shared__ __align__(16) _Float16 lA[64 * 40];   // [m][k]
    __shared__ __align__(16) _Float16 lB[64 * 40];   // [n][k] (transposed)

    const int tid  = threadIdx.x;
    const int lane = tid & 31;
    const int wid  = tid >> 5;
    const int wm   = wid & 3;        // 0..3  -> M sub-tile
    const int wg   = wid >> 2;       // 0..1  -> N sub-tile pair
    const int hi   = lane >> 4;
    const int l15  = lane & 15;

    // ---- hoisted B-staging address math: cc invariant across K-steps ----
    const int cc   = tid & 63;           // column within tile
    const int kkb  = tid >> 6;           // 0..3, kk = kkb + 4*e
    const int colS = n0 + cc;
    const bool inB = colS < L;
    const int bbS  = inB ? (colS / Lb) : 0;
    const int jjS  = colS - bbS * Lb;
    const float* __restrict__ xBase = X + (size_t)bbS * Cin * Lb + jjS;

    // ---- hoisted A-staging address math ----
    const int ar  = tid >> 3;            // 0..31 (rows ar and ar+32)
    const int ac4 = (tid & 7) * 4;       // 0..28

    v8f acc0 = {0.f,0.f,0.f,0.f,0.f,0.f,0.f,0.f};
    v8f acc1 = {0.f,0.f,0.f,0.f,0.f,0.f,0.f,0.f};

    for (int k = 0; k < Cin; k += 32) {
        // ---- stage A: 64 rows x 32 k ----
        {
            const float* wp0 = W + (size_t)(m0 + ar) * Cin + k + ac4;
            const float* wp1 = wp0 + (size_t)32 * Cin;
            float4 w0 = *(const float4*)wp0;
            float4 w1 = *(const float4*)wp1;
            __builtin_prefetch(wp0 + 32, 0, 1);
            __builtin_prefetch(wp1 + 32, 0, 1);
            *(v4h*)(lA + ar * 40 + ac4) =
                (v4h){(_Float16)w0.x, (_Float16)w0.y, (_Float16)w0.z, (_Float16)w0.w};
            *(v4h*)(lA + (ar + 32) * 40 + ac4) =
                (v4h){(_Float16)w1.x, (_Float16)w1.y, (_Float16)w1.z, (_Float16)w1.w};
        }
        // ---- stage B transposed: 32 k x 64 cols (addresses hoisted) ----
        if (inB) {
            const float* xk = xBase + (size_t)k * Lb;
            #pragma unroll
            for (int e = 0; e < 8; ++e) {
                const int kk = kkb + e * 4;
                lB[cc * 40 + kk] = (_Float16)xk[(size_t)kk * Lb];
            }
            __builtin_prefetch(xk + (size_t)32 * Lb, 0, 1);
        } else {
            #pragma unroll
            for (int e = 0; e < 8; ++e)
                lB[cc * 40 + (kkb + e * 4)] = (_Float16)0.f;
        }
        __syncthreads();

        // ---- fragments: one A, two B; two WMMAs sharing A ----
        const int arow = (wm * 16 + l15) * 40;
        v8h a0 = *(const v8h*)(lA + arow + hi * 16);
        v8h a1 = *(const v8h*)(lA + arow + hi * 16 + 8);
        v16h av = __builtin_shufflevector(a0, a1, 0,1,2,3,4,5,6,7,8,9,10,11,12,13,14,15);

        const int brow0 = ((wg * 2 + 0) * 16 + l15) * 40;
        const int brow1 = ((wg * 2 + 1) * 16 + l15) * 40;
        v8h b00 = *(const v8h*)(lB + brow0 + hi * 16);
        v8h b01 = *(const v8h*)(lB + brow0 + hi * 16 + 8);
        v8h b10 = *(const v8h*)(lB + brow1 + hi * 16);
        v8h b11 = *(const v8h*)(lB + brow1 + hi * 16 + 8);
        v16h bv0 = __builtin_shufflevector(b00, b01, 0,1,2,3,4,5,6,7,8,9,10,11,12,13,14,15);
        v16h bv1 = __builtin_shufflevector(b10, b11, 0,1,2,3,4,5,6,7,8,9,10,11,12,13,14,15);

        acc0 = __builtin_amdgcn_wmma_f32_16x16x32_f16(false, av, false, bv0, (short)0, acc0, false, false);
        acc1 = __builtin_amdgcn_wmma_f32_16x16x32_f16(false, av, false, bv1, (short)0, acc1, false, false);

        __syncthreads();
    }

    // ---- epilogue: C/D layout M = r + 8*hi, N = l15 ----
    const int mBase = m0 + wm * 16 + hi * 8;
    #pragma unroll
    for (int nt = 0; nt < 2; ++nt) {
        const int col = n0 + (wg * 2 + nt) * 16 + l15;
        if (col < L) {
            const int bb = col / Lb;
            const int jj = col - bb * Lb;
            const size_t obase = (size_t)bb * Cout * Lb + jj;
            const v8f& acc = nt ? acc1 : acc0;
            #pragma unroll
            for (int r = 0; r < 8; ++r) {
                const int m = mBase + r;
                float v = acc[r] + bias[m];
                const size_t oi = obase + (size_t)m * Lb;
                if (R) v += R[oi];
                if (relu) v = fmaxf(v, 0.f);
                Y[oi] = v;
            }
        }
    }
}

// ---------------------------------------------------------------------------
// Embed: feat[b,o,n] = relu( sum_{i<3} W[o,i]*x[b,n,i] + bias[o] ), o<64, n<1024
// ---------------------------------------------------------------------------
__global__ void embed_kernel(const float* __restrict__ x, const float* __restrict__ W,
                             const float* __restrict__ bias, float* __restrict__ feat) {
    int t = blockIdx.x * blockDim.x + threadIdx.x;
    if (t >= 8 * 64 * 1024) return;
    int nIdx = t & 1023;
    int ch   = (t >> 10) & 63;
    int b    = t >> 16;
    const float* xp = x + ((size_t)b * 1024 + nIdx) * 3;
    float v = W[ch*3+0]*xp[0] + W[ch*3+1]*xp[1] + W[ch*3+2]*xp[2] + bias[ch];
    feat[t] = fmaxf(v, 0.f);
}

// ---------------------------------------------------------------------------
// Farthest point sampling: one block per batch; stable argmax tie-break.
// ---------------------------------------------------------------------------
__global__ __launch_bounds__(256)
void fps_kernel(const float* __restrict__ xyz, int n, int s, int* __restrict__ fi) {
    const int b = blockIdx.x;
    const int tid = threadIdx.x;
    __shared__ float dist[1024];
    __shared__ float rv[256];
    __shared__ int   ri[256];
    __shared__ int   curFar;
    const float* base = xyz + (size_t)b * n * 3;
    for (int i = tid; i < n; i += 256) dist[i] = 1e10f;
    if (tid == 0) curFar = 0;
    __syncthreads();
    for (int it = 0; it < s; ++it) {
        const int far = curFar;
        if (tid == 0) fi[b * s + it] = far;
        const float cx = base[far*3+0], cy = base[far*3+1], cz = base[far*3+2];
        float bm = -1.f; int bi = 0;
        for (int i = tid; i < n; i += 256) {
            float dx = base[i*3+0]-cx, dy = base[i*3+1]-cy, dz = base[i*3+2]-cz;
            float d = dx*dx + dy*dy + dz*dz;
            float nd = fminf(dist[i], d);
            dist[i] = nd;
            if (nd > bm) { bm = nd; bi = i; }   // ascending i -> first index on ties
        }
        rv[tid] = bm; ri[tid] = bi;
        __syncthreads();
        for (int off = 128; off > 0; off >>= 1) {
            if (tid < off) {
                if (rv[tid+off] > rv[tid] ||
                    (rv[tid+off] == rv[tid] && ri[tid+off] < ri[tid])) {
                    rv[tid] = rv[tid+off]; ri[tid] = ri[tid+off];
                }
            }
            __syncthreads();
        }
        if (tid == 0) curFar = ri[0];
        __syncthreads();
    }
}

// ---------------------------------------------------------------------------
// KNN (24 smallest d2, stable) + gather new_xyz. One thread per query.
// ---------------------------------------------------------------------------
__global__ void knn_kernel(const float* __restrict__ xyz, const int* __restrict__ fi,
                           int n, int s, int total, int* __restrict__ ki,
                           float* __restrict__ new_xyz) {
    int t = blockIdx.x * blockDim.x + threadIdx.x;
    if (t >= total) return;
    int b = t / s;
    const float* base = xyz + (size_t)b * n * 3;
    int f = fi[t];
    float qx = base[f*3+0], qy = base[f*3+1], qz = base[f*3+2];
    new_xyz[t*3+0] = qx; new_xyz[t*3+1] = qy; new_xyz[t*3+2] = qz;
    float bd[KNN]; int bidx[KNN];
    for (int j = 0; j < KNN; ++j) { bd[j] = 1e30f; bidx[j] = 0; }
    for (int i = 0; i < n; ++i) {
        float dx = base[i*3+0]-qx, dy = base[i*3+1]-qy, dz = base[i*3+2]-qz;
        float d2 = dx*dx + dy*dy + dz*dz;
        if (d2 < bd[KNN-1]) {
            int j = KNN - 1;
            while (j > 0 && bd[j-1] > d2) { bd[j] = bd[j-1]; bidx[j] = bidx[j-1]; --j; }
            bd[j] = d2; bidx[j] = i;
        }
    }
    for (int j = 0; j < KNN; ++j) ki[t*KNN + j] = bidx[j];
}

// ---------------------------------------------------------------------------
// Per-batch sum / sumsq of (grouped - anchor); deterministic two-pass.
// ---------------------------------------------------------------------------
__global__ __launch_bounds__(256)
void stats_kernel(const float* __restrict__ feat, const int* __restrict__ fi,
                  const int* __restrict__ ki, int C, int n, int s,
                  float* __restrict__ partials) {
    const int b = blockIdx.y;
    const int tid = threadIdx.x;
    const int M = s * KNN * C;
    float s1 = 0.f, s2 = 0.f;
    for (int e = blockIdx.x * 256 + tid; e < M; e += gridDim.x * 256) {
        int q  = e / (KNN * C);
        int r0 = e - q * KNN * C;
        int kk = r0 / C;
        int ch = r0 - kk * C;
        const float* fr = feat + ((size_t)b * C + ch) * n;
        float g = fr[ki[(b * s + q) * KNN + kk]];
        float a = fr[fi[b * s + q]];
        float d = g - a;
        s1 += d; s2 += d * d;
    }
    __shared__ float sh1[256], sh2[256];
    sh1[tid] = s1; sh2[tid] = s2;
    __syncthreads();
    for (int off = 128; off > 0; off >>= 1) {
        if (tid < off) { sh1[tid] += sh1[tid+off]; sh2[tid] += sh2[tid+off]; }
        __syncthreads();
    }
    if (tid == 0) {
        partials[(b * gridDim.x + blockIdx.x) * 2 + 0] = sh1[0];
        partials[(b * gridDim.x + blockIdx.x) * 2 + 1] = sh2[0];
    }
}

__global__ void std_finalize_kernel(const float* __restrict__ partials, int nblk,
                                    int cnt, float* __restrict__ stdv) {
    int b = threadIdx.x;
    if (b >= 8) return;
    float s1 = 0.f, s2 = 0.f;
    for (int i = 0; i < nblk; ++i) {
        s1 += partials[(b * nblk + i) * 2 + 0];
        s2 += partials[(b * nblk + i) * 2 + 1];
    }
    float fc = (float)cnt;
    float var = (s2 - s1 * s1 / fc) / (fc - 1.f);
    stdv[b] = sqrtf(fmaxf(var, 0.f));
}

// ---------------------------------------------------------------------------
// Build transfer input:  Xs[bb, ch, kk] = alpha*norm(g - anchor)+beta,
//                        Xs[bb, C+ch, kk] = anchor    (bb = b*s+q, Lb=24)
// ---------------------------------------------------------------------------
__global__ void build_group_kernel(const float* __restrict__ feat, const int* __restrict__ fi,
                                   const int* __restrict__ ki, const float* __restrict__ alpha,
                                   const float* __restrict__ beta, const float* __restrict__ stdv,
                                   int C, int n, int s, float* __restrict__ Xs) {
    int t = blockIdx.x * blockDim.x + threadIdx.x;
    const int total = 8 * s * KNN * C;
    if (t >= total) return;
    int ch = t % C;  int r0 = t / C;
    int kk = r0 % KNN; int r1 = r0 / KNN;
    int q  = r1 % s;   int b  = r1 / s;
    const float* fr = feat + ((size_t)b * C + ch) * n;
    float anchor = fr[fi[b * s + q]];
    float g      = fr[ki[(b * s + q) * KNN + kk]];
    float inv = 1.f / (stdv[b] + 1e-5f);
    float gn  = alpha[ch] * ((g - anchor) * inv) + beta[ch];
    size_t bb = (size_t)b * s + q;
    Xs[bb * (2 * C * KNN) + (size_t)ch * KNN + kk]       = gn;
    Xs[bb * (2 * C * KNN) + (size_t)(C + ch) * KNN + kk] = anchor;
}

// ---------------------------------------------------------------------------
// Max over k:  featO[b, ch, q] = max_kk Y[b*s+q, ch, kk]
// ---------------------------------------------------------------------------
__global__ void maxk_kernel(const float* __restrict__ Y, int C, int s,
                            float* __restrict__ featO) {
    int t = blockIdx.x * blockDim.x + threadIdx.x;
    if (t >= 8 * C * s) return;
    int q  = t % s;  int r0 = t / s;
    int ch = r0 % C; int b  = r0 / C;
    const float* yp = Y + (((size_t)b * s + q) * C + ch) * KNN;
    float m = yp[0];
    for (int kk = 1; kk < KNN; ++kk) m = fmaxf(m, yp[kk]);
    featO[t] = m;
}

// ---------------------------------------------------------------------------
// Final global max over points: out[b, ch] = max_q feat[b, ch, q]
// ---------------------------------------------------------------------------
__global__ void final_max_kernel(const float* __restrict__ feat, int C, int s,
                                 float* __restrict__ out) {
    int t = blockIdx.x * blockDim.x + threadIdx.x;
    if (t >= 8 * C) return;
    const float* fp = feat + (size_t)t * s;
    float m = fp[0];
    for (int q = 1; q < s; ++q) m = fmaxf(m, fp[q]);
    out[t] = m;
}

// ---------------------------------------------------------------------------
extern "C" void kernel_launch(void* const* d_in, const int* in_sizes, int n_in,
                              void* d_out, int out_size, void* d_ws, size_t ws_size,
                              hipStream_t stream) {
    const float* x      = (const float*)d_in[0];
    const float* emb_b  = (const float*)d_in[1];   // params flattened alphabetically:
    const float* emb_w  = (const float*)d_in[2];   // embed{b,w}, stages[i]{alpha,beta,pos,pre,transfer}
    static const int DIMS[4] = {128, 256, 512, 1024};

    // ---- workspace layout (floats), 64-elt aligned blocks ----
    float* ws = (float*)d_ws;
    size_t off = 0;
    auto alloc = [&](size_t nf) { float* p = ws + off; off += (nf + 63) & ~(size_t)63; return p; };
    float* featA = alloc(8 * 1024 * 64);
    float* featB = alloc(8 * 1024 * 64);
    float* featC = alloc(8 * 1024 * 64);
    float* xyzA  = alloc(8 * 512 * 3);
    float* xyzB  = alloc(8 * 512 * 3);
    int*   fiP   = (int*)alloc(8 * 512);
    int*   kiP   = (int*)alloc(8 * 512 * KNN);
    float* parts = alloc(8 * 64 * 2);
    float* stdv  = alloc(64);
    float* BUF0  = alloc((size_t)8 * 512 * 128 * KNN);  // 12.58M floats each
    float* BUF1  = alloc((size_t)8 * 512 * 128 * KNN);
    float* BUF2  = alloc((size_t)8 * 512 * 128 * KNN);

    // ---- embed ----
    embed_kernel<<<(8 * 64 * 1024 + 255) / 256, 256, 0, stream>>>(x, emb_w, emb_b, featA);

    const float* xyzCur = x;
    int n = 1024, cin = 64;
    int p = 3;
    for (int st = 0; st < 4; ++st) {
        const int s = n / 2, cout = DIMS[st];
        const float* alpha = (const float*)d_in[p + 0];
        const float* beta  = (const float*)d_in[p + 1];
        const float* p0c1b = (const float*)d_in[p + 2],  *p0c1w = (const float*)d_in[p + 3];
        const float* p0c2b = (const float*)d_in[p + 4],  *p0c2w = (const float*)d_in[p + 5];
        const float* p1c1b = (const float*)d_in[p + 6],  *p1c1w = (const float*)d_in[p + 7];
        const float* p1c2b = (const float*)d_in[p + 8],  *p1c2w = (const float*)d_in[p + 9];
        const float* q0c1b = (const float*)d_in[p + 10], *q0c1w = (const float*)d_in[p + 11];
        const float* q0c2b = (const float*)d_in[p + 12], *q0c2w = (const float*)d_in[p + 13];
        const float* q1c1b = (const float*)d_in[p + 14], *q1c1w = (const float*)d_in[p + 15];
        const float* q1c2b = (const float*)d_in[p + 16], *q1c2w = (const float*)d_in[p + 17];
        const float* trb   = (const float*)d_in[p + 18], *trw   = (const float*)d_in[p + 19];

        float* xyzNew = (st & 1) ? xyzB : xyzA;

        // sampling + grouping
        fps_kernel<<<8, 256, 0, stream>>>(xyzCur, n, s, fiP);
        knn_kernel<<<(8 * s + 255) / 256, 256, 0, stream>>>(xyzCur, fiP, n, s, 8 * s, kiP, xyzNew);
        stats_kernel<<<dim3(64, 8), 256, 0, stream>>>(featA, fiP, kiP, cin, n, s, parts);
        std_finalize_kernel<<<1, 8, 0, stream>>>(parts, 64, s * KNN * cin, stdv);
        build_group_kernel<<<(8 * s * KNN * cin + 255) / 256, 256, 0, stream>>>(
            featA, fiP, kiP, alpha, beta, stdv, cin, n, s, BUF0);

        // transfer + two pre residual blocks (Lb = 24 columns per point)
        const int Lk = 8 * s * KNN;
        dim3 gk((Lk + 63) / 64, cout / 64);
        gemm_wmma_kernel<<<gk, 256, 0, stream>>>(trw,   trb,   BUF0, nullptr, BUF1, cout, 2*cin, KNN, 8*s, 1);
        gemm_wmma_kernel<<<gk, 256, 0, stream>>>(q0c1w, q0c1b, BUF1, nullptr, BUF2, cout, cout, KNN, 8*s, 1);
        gemm_wmma_kernel<<<gk, 256, 0, stream>>>(q0c2w, q0c2b, BUF2, BUF1,    BUF0, cout, cout, KNN, 8*s, 1);
        gemm_wmma_kernel<<<gk, 256, 0, stream>>>(q1c1w, q1c1b, BUF0, nullptr, BUF1, cout, cout, KNN, 8*s, 1);
        gemm_wmma_kernel<<<gk, 256, 0, stream>>>(q1c2w, q1c2b, BUF1, BUF0,    BUF2, cout, cout, KNN, 8*s, 1);

        // pool over k -> feat (b, cout, s)
        maxk_kernel<<<(8 * cout * s + 255) / 256, 256, 0, stream>>>(BUF2, cout, s, featA);

        // two pos residual blocks on (8, cout, s)
        const int Lp = 8 * s;
        dim3 gp((Lp + 63) / 64, cout / 64);
        gemm_wmma_kernel<<<gp, 256, 0, stream>>>(p0c1w, p0c1b, featA, nullptr, featB, cout, cout, s, 8, 1);
        gemm_wmma_kernel<<<gp, 256, 0, stream>>>(p0c2w, p0c2b, featB, featA,   featC, cout, cout, s, 8, 1);
        gemm_wmma_kernel<<<gp, 256, 0, stream>>>(p1c1w, p1c1b, featC, nullptr, featB, cout, cout, s, 8, 1);
        gemm_wmma_kernel<<<gp, 256, 0, stream>>>(p1c2w, p1c2b, featB, featC,   featA, cout, cout, s, 8, 1);

        xyzCur = xyzNew;
        n = s; cin = cout; p += 20;
    }

    // final max over remaining 64 points -> (8, 1024)
    final_max_kernel<<<(8 * 1024 + 255) / 256, 256, 0, stream>>>(featA, 1024, 64, (float*)d_out);
    (void)in_sizes; (void)n_in; (void)out_size; (void)ws_size; (void)emb_b;
}
